// SO2SPDPolarLayer_79654463472007
// MI455X (gfx1250) — matrix operand order repaired and checked by reading, not compiled, not run on hardware
//
#include <hip/hip_runtime.h>
#include <math.h>

#define B_   2
#define T_   8192
#define D_   1024
#define M_   (B_*T_)
#define DFF_ 4096

typedef unsigned short u16;
typedef __attribute__((ext_vector_type(16))) __bf16 v16bf;
typedef __attribute__((ext_vector_type(8)))  float  v8f;

union Frag {
  v16bf v;
  u16   u[16];
  uint4 q[2];
};

__device__ __forceinline__ u16 f2bf(float f) {
  unsigned int u = __float_as_uint(f);
  u += 0x7FFFu + ((u >> 16) & 1u);      // round-to-nearest-even
  return (u16)(u >> 16);
}
__device__ __forceinline__ float bf2f(u16 h) {
  return __uint_as_float(((unsigned int)h) << 16);
}
__device__ __forceinline__ float softplusf(float x) {
  return x > 20.f ? x : log1pf(expf(x));
}

// ------------------------------------------------ weight convert + transpose
// WT[n*K + k] = bf16(W[k*N + n]); tiled through LDS so both sides coalesce.
__global__ __launch_bounds__(256) void wconv_kernel(const float* __restrict__ W,
                                                    u16* __restrict__ WT,
                                                    int K, int N) {
  __shared__ u16 tile[32][33];
  int kt = blockIdx.x << 5, nt = blockIdx.y << 5;
  int tx = threadIdx.x & 31, ty = threadIdx.x >> 5;   // 32 x 8
  for (int r = ty; r < 32; r += 8)
    tile[r][tx] = f2bf(W[(size_t)(kt + r) * N + nt + tx]);
  __syncthreads();
  for (int r = ty; r < 32; r += 8)
    WT[(size_t)(nt + r) * K + kt + tx] = tile[tx][r];
}

// ---------------------------------------------------------------- layernorm
__global__ __launch_bounds__(256) void ln_kernel(const float* __restrict__ x,
                                                 const float* __restrict__ g,
                                                 const float* __restrict__ bta,
                                                 u16* __restrict__ out) {
  int row = blockIdx.x, tid = threadIdx.x;
  size_t base = (size_t)row * D_;
  __shared__ float r1[256], r2[256];
  float s = 0.f, s2 = 0.f;
  for (int i = tid; i < D_; i += 256) { float a = x[base + i]; s += a; s2 += a * a; }
  r1[tid] = s; r2[tid] = s2; __syncthreads();
  for (int o = 128; o > 0; o >>= 1) {
    if (tid < o) { r1[tid] += r1[tid + o]; r2[tid] += r2[tid + o]; }
    __syncthreads();
  }
  float mean = r1[0] * (1.f / D_);
  float var  = r2[0] * (1.f / D_) - mean * mean;
  float inv  = rsqrtf(var + 1e-5f);
  for (int i = tid; i < D_; i += 256)
    out[base + i] = f2bf((x[base + i] - mean) * inv * g[i] + bta[i]);
}

// ---------------------------------------------------------------- bf16 WMMA GEMM
// C[M,N] = act(A_bf16[M,K] @ WT_bf16[N,K]^T + bias) (+ residual)
// 8 waves/block; each wave computes a 16x64 strip (4 accumulators sharing one
// A-fragment). All operand loads are contiguous b128.
#define GEMM_NT 4
__global__ __launch_bounds__(256) void gemm_kernel(
    const u16* __restrict__ A, const u16* __restrict__ WT,
    const float* __restrict__ bias, const float* __restrict__ resid,
    float* __restrict__ Cf, u16* __restrict__ Cb,
    int M, int N, int K, int act, int storeT) {
  int wave = threadIdx.x >> 5, lane = threadIdx.x & 31;
  int hi = (lane >> 4) & 1, l15 = lane & 15;
  int m0 = blockIdx.x << 4;
  int n0 = (blockIdx.y << 9) + (wave << 6);           // 8 waves * 64 cols
  int rowA = m0 + l15;
  v8f acc[GEMM_NT];
#pragma unroll
  for (int nt = 0; nt < GEMM_NT; ++nt)
    acc[nt] = (v8f){0.f,0.f,0.f,0.f,0.f,0.f,0.f,0.f};
  const u16* pa0 = A + (size_t)rowA * K + (hi << 4);
  for (int k0 = 0; k0 < K; k0 += 32) {
    Frag a;
    const u16* pa = pa0 + k0;
    a.q[0] = *(const uint4*)(pa);
    a.q[1] = *(const uint4*)(pa + 8);
    __builtin_prefetch(pa + 64, 0, 3);                // global_prefetch_b8
#pragma unroll
    for (int nt = 0; nt < GEMM_NT; ++nt) {
      Frag b;
      const u16* pw = WT + (size_t)(n0 + (nt << 4) + l15) * K + k0 + (hi << 4);
      b.q[0] = *(const uint4*)(pw);
      b.q[1] = *(const uint4*)(pw + 8);
      acc[nt] = __builtin_amdgcn_wmma_f32_16x16x32_bf16(false, a.v, false, b.v,
                                                        (short)0, acc[nt], false, false);
    }
  }
#pragma unroll
  for (int nt = 0; nt < GEMM_NT; ++nt) {
    int ncol = n0 + (nt << 4) + l15;
    float bs = bias ? bias[ncol] : 0.f;
#pragma unroll
    for (int r = 0; r < 8; ++r) {
      int m = m0 + r + (hi << 3);
      float vv = acc[nt][r] + bs;
      if (act == 1) vv = 0.5f * vv * (1.f + erff(vv * 0.70710678118f));  // exact gelu
      if (resid) vv += resid[(size_t)m * N + ncol];
      if (Cf) Cf[(size_t)m * N + ncol] = vv;
      if (Cb) {
        if (storeT) Cb[(size_t)ncol * M + m] = f2bf(vv);   // transposed (for V)
        else        Cb[(size_t)m * N + ncol] = f2bf(vv);
      }
    }
  }
}

// ---------------------------------------------------------------- clock (N=1 dot)
__global__ __launch_bounds__(256) void clock_kernel(const u16* __restrict__ xn,
                                                    const float* __restrict__ cw,
                                                    const float* __restrict__ cb,
                                                    float* __restrict__ clk) {
  int row = blockIdx.x, tid = threadIdx.x;
  size_t base = (size_t)row * D_;
  __shared__ float red[256];
  float s = 0.f;
  for (int i = tid; i < D_; i += 256) s += bf2f(xn[base + i]) * cw[i];
  red[tid] = s; __syncthreads();
  for (int o = 128; o > 0; o >>= 1) { if (tid < o) red[tid] += red[tid + o]; __syncthreads(); }
  if (tid == 0) clk[row] = softplusf(red[0] + cb[0]);
}

// ---------------------------------------------------------------- l2norm + rope
__global__ __launch_bounds__(256) void rowops_kernel(float* __restrict__ q,
                                                     float* __restrict__ k) {
  int row = blockIdx.x, tid = threadIdx.x;
  int t = row & (T_ - 1);
  size_t base = (size_t)row * D_;
  __shared__ float rq[256], rk[256];
  float sq = 0.f, sk = 0.f;
  for (int i = tid; i < D_; i += 256) {
    float a = q[base + i]; sq += a * a;
    float c = k[base + i]; sk += c * c;
  }
  rq[tid] = sq; rk[tid] = sk; __syncthreads();
  for (int o = 128; o > 0; o >>= 1) {
    if (tid < o) { rq[tid] += rq[tid + o]; rk[tid] += rk[tid + o]; }
    __syncthreads();
  }
  float qs = 1.f / fmaxf(sqrtf(rq[0]), 1e-12f);
  float ks = 1.f / fmaxf(sqrtf(rk[0]), 1e-12f);
  const int half = D_ / 2;
  const float lg = 9.21034037198f / (float)half;      // ln(10000)/half
  for (int i = tid; i < half; i += 256) {
    float fr = expf(-(float)i * lg);
    float th = (float)t * fr;
    float c = cosf(th), s = sinf(th);
    float x1 = q[base + 2 * i] * qs, x2 = q[base + 2 * i + 1] * qs;
    q[base + 2 * i]     = x1 * c - x2 * s;
    q[base + 2 * i + 1] = x1 * s + x2 * c;
    x1 = k[base + 2 * i] * ks; x2 = k[base + 2 * i + 1] * ks;
    k[base + 2 * i]     = x1 * c - x2 * s;
    k[base + 2 * i + 1] = x1 * s + x2 * c;
  }
}

// ---------------------------------------------------------------- per-channel scans
__global__ __launch_bounds__(256) void scan_kernel(
    const float* __restrict__ q1, const float* __restrict__ k1,
    const float* __restrict__ graw, const float* __restrict__ praw,
    const float* __restrict__ clk, u16* __restrict__ qb, u16* __restrict__ kb) {
  int ch = blockIdx.x;
  int b = ch / D_, d = ch - b * D_;
  int tid = threadIdx.x;
  __shared__ float sg[256], sp[256];
  float pm = -3.0e38f;
  for (int tt = tid; tt < T_; tt += 256)
    pm = fmaxf(pm, praw[((size_t)(b * T_ + tt)) * D_ + d]);
  sg[tid] = pm; __syncthreads();
  for (int o = 128; o > 0; o >>= 1) { if (tid < o) sg[tid] = fmaxf(sg[tid], sg[tid + o]); __syncthreads(); }
  float pmax = sg[0]; __syncthreads();
  float goff = 0.f, poff = 0.f;
  for (int c = 0; c < T_ / 256; ++c) {
    int tt = (c << 8) + tid;
    size_t idx = ((size_t)(b * T_ + tt)) * D_ + d;
    float gj = -softplusf(graw[idx]);
    float pe = expf(praw[idx] - pmax) * clk[b * T_ + tt];
    sg[tid] = gj; sp[tid] = pe; __syncthreads();
    for (int o = 1; o < 256; o <<= 1) {
      float ag = (tid >= o) ? sg[tid - o] : 0.f;
      float ap = (tid >= o) ? sp[tid - o] : 0.f;
      __syncthreads();
      sg[tid] += ag; sp[tid] += ap; __syncthreads();
    }
    float gcum = goff + sg[tid];
    float pcum = poff + sp[tid];
    float gtot = sg[255], ptot = sp[255];
    __syncthreads();
    goff += gtot; poff += ptot;
    float G = expf(fminf(fmaxf(gcum, -60.f), 50.f));
    qb[idx] = f2bf(q1[idx] * G / (pcum + 1e-8f));
    kb[idx] = f2bf(k1[idx] / (G + 1e-8f) * pe);
  }
}

// ---------------------------------------------------------------- flash attention
// 16-query tile per block, 8 waves; wave w owns DH slice [w*128, w*128+128).
// Vt is transposed [D][B*T] so P x V B-fragments load contiguously.
__global__ __launch_bounds__(256) void attn_kernel(const u16* __restrict__ Q,
                                                   const u16* __restrict__ Kt,
                                                   const u16* __restrict__ Vt,
                                                   u16* __restrict__ O) {
  const float scale = 0.03125f;  // 1/sqrt(1024)
  int b = blockIdx.y;
  int q0 = blockIdx.x << 4;
  int tid = threadIdx.x;
  int wave = tid >> 5, lane = tid & 31;
  int hi = (lane >> 4) & 1, l15 = lane & 15;
  int dbase = wave << 7;
  __shared__ float sS[16 * 32];
  __shared__ u16   sP[16 * 32];
  __shared__ float sM[16], sL[16], sA[16];
  if (tid < 16) { sM[tid] = -3.0e38f; sL[tid] = 0.f; }
  Frag aq[4];
  {
    const u16* pq = Q + ((size_t)(b * T_) + q0 + l15) * D_ + dbase + (hi << 4);
#pragma unroll
    for (int kk = 0; kk < 4; ++kk) {
      aq[kk].q[0] = *(const uint4*)(pq + kk * 32);
      aq[kk].q[1] = *(const uint4*)(pq + kk * 32 + 8);
    }
  }
  v8f oacc[8];
#pragma unroll
  for (int nt = 0; nt < 8; ++nt) oacc[nt] = (v8f){0.f,0.f,0.f,0.f,0.f,0.f,0.f,0.f};
  __syncthreads();

  for (int j0 = 0; j0 <= q0 + 15; j0 += 32) {
    sS[tid] = 0.f; sS[tid + 256] = 0.f;
    __syncthreads();
    // ---- scores: partial over this wave's 128-wide feature slice
#pragma unroll
    for (int sub = 0; sub < 2; ++sub) {
      int key = j0 + (sub << 4) + l15;
      if (key > T_ - 1) key = T_ - 1;                  // clamp; masked below
      v8f sacc = (v8f){0.f,0.f,0.f,0.f,0.f,0.f,0.f,0.f};
      const u16* pk = Kt + ((size_t)(b * T_) + key) * D_ + dbase + (hi << 4);
#pragma unroll
      for (int kk = 0; kk < 4; ++kk) {
        const u16* pkk = pk + kk * 32;
        Frag bk;
        bk.q[0] = *(const uint4*)(pkk);
        bk.q[1] = *(const uint4*)(pkk + 8);
        sacc = __builtin_amdgcn_wmma_f32_16x16x32_bf16(false, aq[kk].v, false, bk.v,
                                                       (short)0, sacc, false, false);
      }
#pragma unroll
      for (int r = 0; r < 8; ++r)
        atomicAdd(&sS[(r + (hi << 3)) * 32 + (sub << 4) + l15], sacc[r]);
    }
    __syncthreads();
    // ---- online softmax (one thread per query row)
    if (tid < 16) {
      int m = tid, qrow = q0 + m;
      float mx = sM[m], mnew = mx;
      float sc[32];
#pragma unroll
      for (int n = 0; n < 32; ++n) {
        int key = j0 + n;
        float v = (key <= qrow) ? sS[m * 32 + n] * scale : -3.0e38f;
        sc[n] = v;
        mnew = fmaxf(mnew, v);
      }
      float alpha = expf(mx - mnew);
      float lsum = sL[m] * alpha;
#pragma unroll
      for (int n = 0; n < 32; ++n) {
        float p = (sc[n] > -1.0e38f) ? expf(sc[n] - mnew) : 0.f;
        sP[m * 32 + n] = f2bf(p);
        lsum += p;
      }
      sM[m] = mnew; sL[m] = lsum; sA[m] = alpha;
    }
    __syncthreads();
    // ---- rescale accumulators, then P x V over this wave's slice
    Frag ap;
    {
      const u16* pp = &sP[l15 * 32 + (hi << 4)];
      ap.q[0] = *(const uint4*)(pp);
      ap.q[1] = *(const uint4*)(pp + 8);
    }
#pragma unroll
    for (int nt = 0; nt < 8; ++nt) {
#pragma unroll
      for (int r = 0; r < 8; ++r) oacc[nt][r] *= sA[r + (hi << 3)];
    }
    int key0 = j0 + (hi << 4);
    bool inb = (key0 + 15) < T_;
#pragma unroll
    for (int nt = 0; nt < 8; ++nt) {
      Frag bv;
      const u16* pv = Vt + (size_t)(dbase + (nt << 4) + l15) * M_ + (size_t)b * T_;
      if (inb) {
        const u16* p = pv + key0;
        bv.q[0] = *(const uint4*)(p);
        bv.q[1] = *(const uint4*)(p + 8);
      } else {
#pragma unroll
        for (int h = 0; h < 16; ++h) {
          int key = key0 + h;
          if (key > T_ - 1) key = T_ - 1;              // clamped row; P==0 there
          bv.u[h] = pv[key];
        }
      }
      oacc[nt] = __builtin_amdgcn_wmma_f32_16x16x32_bf16(false, ap.v, false, bv.v,
                                                         (short)0, oacc[nt], false, false);
    }
    __syncthreads();
  }
  // ---- epilogue: normalize by row sums, store bf16 (row-major, A-operand-ready)
#pragma unroll
  for (int nt = 0; nt < 8; ++nt) {
#pragma unroll
    for (int r = 0; r < 8; ++r) {
      int m = r + (hi << 3);
      float l = sL[m];
      float val = oacc[nt][r] / (l > 0.f ? l : 1.f);
      O[((size_t)(b * T_) + q0 + m) * D_ + dbase + (nt << 4) + l15] = f2bf(val);
    }
  }
}

// ---------------------------------------------------------------- launch
extern "C" void kernel_launch(void* const* d_in, const int* in_sizes, int n_in,
                              void* d_out, int out_size, void* d_ws, size_t ws_size,
                              hipStream_t stream) {
  (void)in_sizes; (void)n_in; (void)out_size; (void)ws_size;
  const float* x       = (const float*)d_in[0];
  const float* Wq      = (const float*)d_in[1];
  const float* Wk      = (const float*)d_in[2];
  const float* Wv      = (const float*)d_in[3];
  const float* gate_W  = (const float*)d_in[4];
  const float* gate_b  = (const float*)d_in[5];
  const float* p_W     = (const float*)d_in[6];
  const float* p_b     = (const float*)d_in[7];
  const float* clock_W = (const float*)d_in[8];
  const float* clock_b = (const float*)d_in[9];
  const float* out_W   = (const float*)d_in[10];
  const float* out_b   = (const float*)d_in[11];
  const float* ln1_g   = (const float*)d_in[12];
  const float* ln1_b   = (const float*)d_in[13];
  const float* ln2_g   = (const float*)d_in[14];
  const float* ln2_b   = (const float*)d_in[15];
  const float* mlp_W1  = (const float*)d_in[16];
  const float* mlp_b1  = (const float*)d_in[17];
  const float* mlp_W2  = (const float*)d_in[18];
  const float* mlp_b2  = (const float*)d_in[19];

  const size_t MB = 1024ull * 1024ull;
  char* ws = (char*)d_ws;
  u16*   xn_bf  = (u16*)(ws + 0);           // 32 MB
  float* qraw   = (float*)(ws + 32 * MB);   // 64 MB
  float* kraw   = (float*)(ws + 96 * MB);   // 64 MB
  float* graw   = (float*)(ws + 160 * MB);  // 64 MB
  float* praw   = (float*)(ws + 224 * MB);  // 64 MB
  u16*   vt     = (u16*)(ws + 288 * MB);    // 32 MB  (V transposed [D][B*T])
  u16*   qbf    = (u16*)(ws + 320 * MB);    // 32 MB
  u16*   kbf    = (u16*)(ws + 352 * MB);    // 32 MB
  float* clockf = (float*)(ws + 384 * MB);  // 64 KB
  // pre-converted bf16 [N,K] weights:
  u16*   wqT = (u16*)(ws + 385 * MB);       // 2 MB each (DxD)
  u16*   wkT = (u16*)(ws + 387 * MB);
  u16*   wvT = (u16*)(ws + 389 * MB);
  u16*   wgT = (u16*)(ws + 391 * MB);
  u16*   wpT = (u16*)(ws + 393 * MB);
  u16*   woT = (u16*)(ws + 395 * MB);
  u16*   w1T = (u16*)(ws + 397 * MB);       // 8 MB
  u16*   w2T = (u16*)(ws + 405 * MB);       // 8 MB
  // reuse dead regions:
  u16*   attnbf = (u16*)qraw;               // qraw dead after scan
  float* x2     = kraw;                     // kraw dead after scan
  u16*   hbf    = xn_bf;                    // xn dead after projections
  u16*   mlpmid = (u16*)graw;               // graw+praw dead after scan (128 MB)

  dim3 blk(256);
  dim3 gD(M_ / 16, D_ / 512);
  dim3 gF(M_ / 16, DFF_ / 512);

  // weight prep (bf16, transposed)
  wconv_kernel<<<dim3(D_ / 32, D_ / 32), blk, 0, stream>>>(Wq, wqT, D_, D_);
  wconv_kernel<<<dim3(D_ / 32, D_ / 32), blk, 0, stream>>>(Wk, wkT, D_, D_);
  wconv_kernel<<<dim3(D_ / 32, D_ / 32), blk, 0, stream>>>(Wv, wvT, D_, D_);
  wconv_kernel<<<dim3(D_ / 32, D_ / 32), blk, 0, stream>>>(gate_W, wgT, D_, D_);
  wconv_kernel<<<dim3(D_ / 32, D_ / 32), blk, 0, stream>>>(p_W, wpT, D_, D_);
  wconv_kernel<<<dim3(D_ / 32, D_ / 32), blk, 0, stream>>>(out_W, woT, D_, D_);
  wconv_kernel<<<dim3(D_ / 32, DFF_ / 32), blk, 0, stream>>>(mlp_W1, w1T, D_, DFF_);
  wconv_kernel<<<dim3(DFF_ / 32, D_ / 32), blk, 0, stream>>>(mlp_W2, w2T, DFF_, D_);

  ln_kernel<<<M_, blk, 0, stream>>>(x, ln1_g, ln1_b, xn_bf);
  gemm_kernel<<<gD, blk, 0, stream>>>(xn_bf, wqT, nullptr, nullptr, qraw, nullptr, M_, D_, D_, 0, 0);
  gemm_kernel<<<gD, blk, 0, stream>>>(xn_bf, wkT, nullptr, nullptr, kraw, nullptr, M_, D_, D_, 0, 0);
  gemm_kernel<<<gD, blk, 0, stream>>>(xn_bf, wvT, nullptr, nullptr, nullptr, vt, M_, D_, D_, 0, 1);
  gemm_kernel<<<gD, blk, 0, stream>>>(xn_bf, wgT, gate_b, nullptr, graw, nullptr, M_, D_, D_, 0, 0);
  gemm_kernel<<<gD, blk, 0, stream>>>(xn_bf, wpT, p_b, nullptr, praw, nullptr, M_, D_, D_, 0, 0);
  clock_kernel<<<M_, blk, 0, stream>>>(xn_bf, clock_W, clock_b, clockf);
  rowops_kernel<<<M_, blk, 0, stream>>>(qraw, kraw);
  scan_kernel<<<B_ * D_, blk, 0, stream>>>(qraw, kraw, graw, praw, clockf, qbf, kbf);
  attn_kernel<<<dim3(T_ / 16, B_), blk, 0, stream>>>(qbf, kbf, vt, attnbf);
  gemm_kernel<<<gD, blk, 0, stream>>>(attnbf, woT, out_b, x, x2, nullptr, M_, D_, D_, 0, 0);
  ln_kernel<<<M_, blk, 0, stream>>>(x2, ln2_g, ln2_b, hbf);
  gemm_kernel<<<gF, blk, 0, stream>>>(hbf, w1T, mlp_b1, nullptr, nullptr, mlpmid, M_, DFF_, D_, 1, 0);
  gemm_kernel<<<gD, blk, 0, stream>>>(mlpmid, w2T, mlp_b2, x2, (float*)d_out, nullptr, M_, D_, DFF_, 0, 0);
}
